// Model_81990925681019
// MI455X (gfx1250) — matrix operand order repaired and checked by reading, not compiled, not run on hardware
//
#include <hip/hip_runtime.h>
#include <hip/hip_bf16.h>
#include <math.h>

typedef __attribute__((ext_vector_type(16))) __bf16 v16bf;
typedef __attribute__((ext_vector_type(8)))  float  v8f;
typedef __attribute__((ext_vector_type(4)))  unsigned v4u;
typedef __attribute__((ext_vector_type(8)))  unsigned v8u;
typedef unsigned short bf16_t;

static constexpr int cB = 16, cL = 2048, cD = 256, cA = 128, cN = 256, cH = 128, cC = 128;
static constexpr int cNY = 6135;           // 2046 + 2045 + 2044
static constexpr float cEPS = 1e-5f;

// ---------- bf16 helpers (bit-level, no __bf16 scalar arithmetic) ----------
__device__ __forceinline__ bf16_t f2bf(float f) {
  unsigned u = __float_as_uint(f);
  u += 0x7FFFu + ((u >> 16) & 1u);         // round-to-nearest-even
  return (bf16_t)(u >> 16);
}
__device__ __forceinline__ float bf2f(bf16_t h) {
  return __uint_as_float(((unsigned)h) << 16);
}

__device__ __forceinline__ void wait_asynccnt0() {
#if __has_builtin(__builtin_amdgcn_s_wait_asynccnt)
  __builtin_amdgcn_s_wait_asynccnt(0);
#else
  asm volatile("s_wait_asynccnt 0x0" ::: "memory");
#endif
}

// ---------- WMMA fragment loaders (layouts per CDNA5 ISA 7.12.2) ----------
// A 16x32 bf16: lanes 0-15 rows M=0..15 hold K={0..7,16..23}; lanes 16-31 hold K={8..15,24..31}
__device__ __forceinline__ v16bf fragA(const bf16_t* rowp, int half_id) {
  union { v16bf v; unsigned u[8]; } r;
  const unsigned* p = (const unsigned*)rowp;   // 32 bf16 = 16 dwords, pair index = k/2
#pragma unroll
  for (int pp = 0; pp < 8; ++pp) {
    int idx = pp + 4 * half_id + ((pp >= 4) ? 4 : 0);
    r.u[pp] = p[idx];
  }
  return r.v;
}
// B 32x16 bf16: lanes 0-15 col N hold K={0..7,16..23}; lanes 16-31 hold K={8..15,24..31}
__device__ __forceinline__ v16bf fragB(const bf16_t* rowp, int half_id) {
  union { v16bf v; unsigned u[8]; } r;
  const unsigned* p = (const unsigned*)rowp;
#pragma unroll
  for (int pp = 0; pp < 8; ++pp) r.u[pp] = p[pp + 4 * half_id + ((pp >= 4) ? 4 : 0)];
  return r.v;
}

// ---------- Generic bf16 WMMA GEMM: C = epi(beta*C + rowScale[m]*(A*B) + bias[n]) ----------
// A element (m,k) = A[sAb*z + m*sAm + k*sAk]   (bf16)
// B element (n,k) = B[n*sBn + k*sBk]           (bf16)  -> i.e. computes A @ B^T
// C element (m,n) = C[sCb*z + m*ldc + n]       (f32 or bf16 via Cb)
struct GemmP {
  const bf16_t* A; long long sAm, sAk, sAb;
  const bf16_t* B; long long sBn, sBk;
  float* C; bf16_t* Cb; long long ldc, sCb;
  const float* bias; long long sBias;
  const float* rowScale; long long sRS;
  const float* postScale; long long sPS;
  int M, N, K; float beta;
};

template <int ACT, int OBF>   // ACT: 0=none 1=relu 2=tanh
__global__ __launch_bounds__(256) void k_gemm(GemmP p) {
  __shared__ __align__(16) bf16_t As[128 * 32];
  __shared__ __align__(16) bf16_t Bs[64 * 32];

  const int tid = threadIdx.x, lane = tid & 31, wave = tid >> 5;
  const int half_id = lane >> 4, mr = lane & 15;
  const int wr = wave >> 1, wc = wave & 1;          // 4x2 wave grid -> 128x64 C tile
  const int rowBase = blockIdx.y * 128;
  const int colBase = blockIdx.x * 64;
  const long long zb = blockIdx.z;
  const bf16_t* A = p.A + zb * p.sAb;

  const int arow = tid >> 1, acol = (tid & 1) * 16;  // A: 128 rows x 32, 16 elems/thread
  const int brow = tid >> 2, bcol = (tid & 3) * 8;   // B: 64 rows x 32,  8 elems/thread

  // LDS byte offsets (generic shared addr low 32 bits == group-relative LDS offset)
  const unsigned ldsAs = (unsigned)(size_t)(void*)As;
  const unsigned ldsBs = (unsigned)(size_t)(void*)Bs;
  const bool tdmB = (p.sBk == 1) && (colBase + 64 <= p.N);   // full, contiguous-K weight tile

  v8f acc[2][2] = {};

  for (int k0 = 0; k0 < p.K; k0 += 32) {
    {  // ---- stage A tile: ASYNCcnt path (no VGPR round-trip) when contiguous ----
      int gm = rowBase + arow;
      bf16_t* dst = &As[arow * 32 + acol];
      if (gm < p.M) {
        const bf16_t* src = A + (long long)gm * p.sAm + (long long)(k0 + acol) * p.sAk;
        if (p.sAk == 1 && (k0 + acol + 16) <= p.K) {
          unsigned l0 = ldsAs + (unsigned)((arow * 32 + acol) * 2);
          unsigned long long ga = (unsigned long long)src;
          asm volatile("global_load_async_to_lds_b128 %0, %1, off"
                       :: "v"(l0), "v"(ga) : "memory");
          asm volatile("global_load_async_to_lds_b128 %0, %1, off"
                       :: "v"(l0 + 16u), "v"(ga + 16ull) : "memory");
          if (k0 + 32 < p.K) __builtin_prefetch((const void*)(src + 32), 0, 1);
        } else {
#pragma unroll 4
          for (int c = 0; c < 16; ++c) {
            int kk = k0 + acol + c;
            dst[c] = (kk < p.K) ? src[(long long)c * p.sAk] : (bf16_t)0;
          }
        }
      } else {
#pragma unroll
        for (int c = 0; c < 16; ++c) dst[c] = 0;
      }
    }
    // ---- stage B tile: TDM (tensor DMA, TENSORcnt) for weight tiles ----
    if (tdmB) {
      if (wave == 0) {
        unsigned long long ga = (unsigned long long)(p.B + (long long)colBase * p.sBn + k0);
        v4u g0; v8u g1;
        g0.x = 1u;                                            // count=1, user mode
        g0.y = ldsBs;                                         // lds_addr (bytes)
        g0.z = (unsigned)(ga & 0xffffffffull);                // global_addr[31:0]
        g0.w = (unsigned)((ga >> 32) & 0x01ffffffull) | (2u << 30);  // addr[56:32] | type=2
        g1[0] = 1u << 16;                                     // data_size = 2 bytes
        g1[1] = 32u << 16;                                    // tensor_dim0 = 32
        g1[2] = 64u << 16;                                    // tensor_dim1 = 64
        g1[3] = 32u << 16;                                    // tile_dim0 = 32
        g1[4] = 64u;                                          // tile_dim1 = 64, tile_dim2 = 0
        g1[5] = (unsigned)p.sBn;                              // tensor_dim0_stride (elements)
        g1[6] = 0u; g1[7] = 0u;
        asm volatile("tensor_load_to_lds %0, %1" :: "s"(g0), "s"(g1) : "memory");
      }
    } else {
      int gn = colBase + brow;
      bf16_t* dst = &Bs[brow * 32 + bcol];
      if (gn < p.N) {
        const bf16_t* src = p.B + (long long)gn * p.sBn + (long long)(k0 + bcol) * p.sBk;
#pragma unroll 4
        for (int c = 0; c < 8; ++c) {
          int kk = k0 + bcol + c;
          dst[c] = (kk < p.K) ? src[(long long)c * p.sBk] : (bf16_t)0;
        }
      } else {
#pragma unroll
        for (int c = 0; c < 8; ++c) dst[c] = 0;
      }
    }
    wait_asynccnt0();                        // own wave's async LDS writes visible
    if (tdmB && wave == 0) __builtin_amdgcn_s_wait_tensorcnt(0);
    __syncthreads();

    v16bf af[2], bw[2];
    af[0] = fragA(&As[(wr * 32 + mr) * 32], half_id);
    af[1] = fragA(&As[(wr * 32 + 16 + mr) * 32], half_id);
    bw[0] = fragB(&Bs[(wc * 32 + mr) * 32], half_id);
    bw[1] = fragB(&Bs[(wc * 32 + 16 + mr) * 32], half_id);
#pragma unroll
    for (int i = 0; i < 2; ++i)
#pragma unroll
      for (int j = 0; j < 2; ++j)
        acc[i][j] = __builtin_amdgcn_wmma_f32_16x16x32_bf16(
            false, af[i], false, bw[j], (short)0, acc[i][j], false, false);
    __syncthreads();
  }

  // epilogue
  const long long cb = zb * p.sCb;
  const float* bias = p.bias ? p.bias + zb * p.sBias : nullptr;
  const float* rsp = p.rowScale ? p.rowScale + zb * p.sRS : nullptr;
  const float* psp = p.postScale ? p.postScale + zb * p.sPS : nullptr;
#pragma unroll
  for (int i = 0; i < 2; ++i)
#pragma unroll
    for (int j = 0; j < 2; ++j) {
      int n = colBase + wc * 32 + j * 16 + mr;
      if (n >= p.N) continue;
      float bv = bias ? bias[n] : 0.f;
#pragma unroll
      for (int r = 0; r < 8; ++r) {
        int m = rowBase + wr * 32 + i * 16 + r + 8 * half_id;
        if (m >= p.M) continue;
        float v = acc[i][j][r];
        if (rsp) v *= rsp[m];
        v += bv;
        if (p.beta != 0.f) v += p.beta * p.C[cb + (long long)m * p.ldc + n];
        if (ACT == 1) v = fmaxf(v, 0.f);
        else if (ACT == 2) v = tanhf(v);
        if (psp) v *= psp[m];
        if (OBF) p.Cb[cb + (long long)m * p.ldc + n] = f2bf(v);
        else     p.C [cb + (long long)m * p.ldc + n] = v;
      }
    }
}

// ---------- small kernels ----------
__global__ void k_f2bf(const float* in, bf16_t* out, long long n) {
  long long i = (long long)blockIdx.x * 256 + threadIdx.x;
  if (i < n) out[i] = f2bf(in[i]);
}

__global__ void k_embed(const int* x, const float* emb, bf16_t* out, long long n) {
  long long i = (long long)blockIdx.x * 256 + threadIdx.x;
  if (i >= n) return;
  long long t = i / cD; int d = (int)(i % cD);
  out[i] = f2bf(emb[(long long)x[t] * cD + d]);
}

__global__ void k_fm(const int* nodes, const float* mask, const float* emb, bf16_t* out) {
  long long i = (long long)blockIdx.x * 256 + threadIdx.x;
  if (i >= (long long)cB * cN * cD) return;
  long long bn = i / cD; int d = (int)(i % cD);
  out[i] = f2bf(emb[(long long)nodes[bn] * cD + d] * mask[bn]);
}

__global__ void k_convw(const float* w, bf16_t* out, int KK) {
  int i = blockIdx.x * 256 + threadIdx.x;
  int tot = cD * cD * KK;
  if (i >= tot) return;
  int o = i / (cD * KK), r = i % (cD * KK), ii = r / KK, kk = r % KK;
  out[(long long)o * (KK * cD) + kk * cD + ii] = f2bf(w[i]);
}

__global__ void k_ddsums(const float* DDm, float* invRow, float* invCol) {
  int a = threadIdx.x;   // 128 threads
  float sr = 0.f, sc = 0.f;
  for (int j = 0; j < cA; ++j) { sr += DDm[a * cA + j]; sc += DDm[j * cA + a]; }
  invRow[a] = 1.f / (sr + cEPS);
  invCol[a] = 1.f / (sc + cEPS);
}

__global__ void k_dfsum(const float* DF, float* inv) {
  int i = blockIdx.x * 256 + threadIdx.x;
  if (i >= cB * cN) return;
  int b = i / cN, n = i % cN;
  const float* p = DF + (long long)b * cA * cN + n;
  float s = 0.f;
  for (int a = 0; a < cA; ++a) s += p[(long long)a * cN];
  inv[i] = 1.f / (s + cEPS);
}

__global__ void k_gru(const float* gi, const float* whh, const float* bhh,
                      float* hout, int reverse) {
  __shared__ float h[cH];
  int n = threadIdx.x, b = blockIdx.x;
  h[n] = 0.f; __syncthreads();
  const float* wr = whh + (long long)n * cH;
  const float* wz = whh + (long long)(cH + n) * cH;
  const float* wn = whh + (long long)(2 * cH + n) * cH;
  float br = bhh[n], bz = bhh[cH + n], bn = bhh[2 * cH + n];
  for (int s = 0; s < cL; ++s) {
    int t = reverse ? (cL - 1 - s) : s;
    const float* g = gi + ((long long)b * cL + t) * (3 * cH);
    float hr = br, hz = bz, hn = bn;
    for (int k = 0; k < cH; ++k) {
      float hk = h[k];
      hr += wr[k] * hk; hz += wz[k] * hk; hn += wn[k] * hk;
    }
    float r  = 1.f / (1.f + expf(-(g[n] + hr)));
    float z  = 1.f / (1.f + expf(-(g[cH + n] + hz)));
    float nn = tanhf(g[2 * cH + n] + r * hn);
    float hnew = (1.f - z) * nn + z * h[n];
    __syncthreads(); h[n] = hnew; __syncthreads();
    hout[((long long)b * cL + t) * cH + n] = hnew;
  }
}

__global__ void k_h1(const float* hf, const float* hb, float* H1, bf16_t* H1b) {
  int i = blockIdx.x * 256 + threadIdx.x;
  if (i >= cB * cD) return;
  int b = i / cD, d = i % cD;
  const float* p = (d < cH) ? (hf + (long long)b * cL * cH + d)
                            : (hb + (long long)b * cL * cH + (d - cH));
  float s = 0.f;
  for (int t = 0; t < cL; ++t) s += p[(long long)t * cH];
  s /= (float)cL;
  H1[i] = s; H1b[i] = f2bf(s);
}

__global__ void k_dotv(const float* U, const float* V, float* s, int nrows) {
  long long i = (long long)blockIdx.x * 256 + threadIdx.x;
  if (i >= (long long)cB * nrows) return;
  const float* u = U + i * cD;
  float acc = 0.f;
  for (int d = 0; d < cD; ++d) acc += u[d] * V[d];
  s[i] = acc;
}

__global__ void k_softmax(float* s, int n) {  // grid = B, block = 256, in-place
  __shared__ float red[256];
  int b = blockIdx.x, t = threadIdx.x;
  float* row = s + (long long)b * n;
  float m = -1e30f;
  for (int j = t; j < n; j += 256) m = fmaxf(m, row[j]);
  red[t] = m; __syncthreads();
  for (int o = 128; o > 0; o >>= 1) { if (t < o) red[t] = fmaxf(red[t], red[t + o]); __syncthreads(); }
  m = red[0]; __syncthreads();
  float sum = 0.f;
  for (int j = t; j < n; j += 256) { float e = expf(row[j] - m); row[j] = e; sum += e; }
  red[t] = sum; __syncthreads();
  for (int o = 128; o > 0; o >>= 1) { if (t < o) red[t] += red[t + o]; __syncthreads(); }
  float inv = 1.f / red[0];
  for (int j = t; j < n; j += 256) row[j] *= inv;
}

__global__ void k_wsum(const float* alpha, const float* U, float* out, int n) {
  int i = blockIdx.x * 256 + threadIdx.x;
  if (i >= cB * cD) return;
  int b = i / cD, d = i % cD;
  const float* a = alpha + (long long)b * n;
  const float* u = U + (long long)b * n * cD + d;
  float s = 0.f;
  for (int j = 0; j < n; ++j) s += a[j] * u[(long long)j * cD];
  out[i] = s;
}

__global__ void k_hcat(const float* Hf, const float* Ht, bf16_t* out) {
  int i = blockIdx.x * 256 + threadIdx.x;
  if (i >= cB * 2 * cD) return;
  int b = i / (2 * cD), d = i % (2 * cD);
  float v = (d < cD) ? Hf[b * cD + d] : Ht[b * cD + d - cD];
  out[i] = f2bf(v);
}

// ---------- host-side helpers ----------
static void launch_gemm(hipStream_t st, int act, int obf, const GemmP& p, int batch) {
  dim3 g((p.N + 63) / 64, (p.M + 127) / 128, batch);
  dim3 b(256, 1, 1);
  if (!obf) {
    if (act == 0)      hipLaunchKernelGGL((k_gemm<0, 0>), g, b, 0, st, p);
    else if (act == 1) hipLaunchKernelGGL((k_gemm<1, 0>), g, b, 0, st, p);
    else               hipLaunchKernelGGL((k_gemm<2, 0>), g, b, 0, st, p);
  } else {
    if (act == 0)      hipLaunchKernelGGL((k_gemm<0, 1>), g, b, 0, st, p);
    else if (act == 1) hipLaunchKernelGGL((k_gemm<1, 1>), g, b, 0, st, p);
    else               hipLaunchKernelGGL((k_gemm<2, 1>), g, b, 0, st, p);
  }
}

extern "C" void kernel_launch(void* const* d_in, const int* in_sizes, int n_in,
                              void* d_out, int out_size, void* d_ws, size_t ws_size,
                              hipStream_t stream) {
  (void)in_sizes; (void)n_in; (void)out_size; (void)ws_size;
  const int*   x     = (const int*)d_in[0];
  const int*   nodes = (const int*)d_in[2];
  const float* nmask = (const float*)d_in[3];
  const float* DDin  = (const float*)d_in[4];
  const float* DFin  = (const float*)d_in[5];
  const float* wemb  = (const float*)d_in[6];
  const float* lemb  = (const float*)d_in[7];
  const float* eemb  = (const float*)d_in[8];
  const float* W1 = (const float*)d_in[9],  * W2 = (const float*)d_in[10];
  const float* W3 = (const float*)d_in[11], * W4 = (const float*)d_in[12];
  const float* W6w = (const float*)d_in[13], * W6bias = (const float*)d_in[14];
  const float* V1 = (const float*)d_in[15], * V2 = (const float*)d_in[16];
  const float* wihf = (const float*)d_in[17], * whhf = (const float*)d_in[18];
  const float* bihf = (const float*)d_in[19], * bhhf = (const float*)d_in[20];
  const float* wihb = (const float*)d_in[21], * whhb = (const float*)d_in[22];
  const float* bihb = (const float*)d_in[23], * bhhb = (const float*)d_in[24];
  const float* c1w = (const float*)d_in[25], * c1b = (const float*)d_in[26];
  const float* c2w = (const float*)d_in[27], * c2b = (const float*)d_in[28];
  const float* c3w = (const float*)d_in[29], * c3b = (const float*)d_in[30];
  const float* cls1w = (const float*)d_in[31], * cls1b = (const float*)d_in[32];
  const float* cls2w = (const float*)d_in[33], * cls2b = (const float*)d_in[34];

  char* wsb = (char*)d_ws; size_t off = 0;
  auto alloc = [&](size_t bytes) -> void* {
    off = (off + 255) & ~(size_t)255;
    void* p = wsb + off; off += bytes; return p;
  };

  bf16_t* xe_b = (bf16_t*)alloc((size_t)cB * cL * cD * 2);
  float*  gi_f = (float*)alloc((size_t)cB * cL * 3 * cH * 4);
  float*  gi_b = (float*)alloc((size_t)cB * cL * 3 * cH * 4);
  float*  hf   = (float*)alloc((size_t)cB * cL * cH * 4);
  float*  hb   = (float*)alloc((size_t)cB * cL * cH * 4);
  float*  H1f  = (float*)alloc(cB * cD * 4);
  bf16_t* H1b  = (bf16_t*)alloc(cB * cD * 2);
  bf16_t* wihf_b = (bf16_t*)alloc(3 * cH * cD * 2);
  bf16_t* wihb_b = (bf16_t*)alloc(3 * cH * cD * 2);
  bf16_t* W1b = (bf16_t*)alloc(cD * cD * 2);
  bf16_t* W2b = (bf16_t*)alloc(cD * cD * 2);
  bf16_t* W3b = (bf16_t*)alloc(cD * cD * 2);
  bf16_t* W4b = (bf16_t*)alloc(cD * cD * 2);
  bf16_t* W6b = (bf16_t*)alloc(cD * 2 * cD * 2);
  bf16_t* Dmb = (bf16_t*)alloc(cA * cD * 2);
  bf16_t* DDb = (bf16_t*)alloc(cA * cA * 2);
  bf16_t* DFb = (bf16_t*)alloc((size_t)cB * cA * cN * 2);
  float* invRow = (float*)alloc(cA * 4);
  float* invCol = (float*)alloc(cA * 4);
  float* invDF  = (float*)alloc(cB * cN * 4);
  float*  P2  = (float*)alloc(cA * cD * 4);
  bf16_t* P2b = (bf16_t*)alloc(cA * cD * 2);
  float*  P3  = (float*)alloc(cA * cD * 4);
  bf16_t* P3b = (bf16_t*)alloc(cA * cD * 2);
  float*  Dhat  = (float*)alloc(cA * cD * 4);
  bf16_t* Dhatb = (bf16_t*)alloc(cA * cD * 2);
  bf16_t* Fmb   = (bf16_t*)alloc((size_t)cB * cN * cD * 2);
  float*  Fhat  = (float*)alloc((size_t)cB * cN * cD * 4);
  bf16_t* Fhatb = (bf16_t*)alloc((size_t)cB * cN * cD * 2);
  float*  ctx1  = (float*)alloc(cB * cD * 4);
  float*  U1    = (float*)alloc((size_t)cB * cN * cD * 4);
  float*  s1    = (float*)alloc(cB * cN * 4);
  float*  Hfv   = (float*)alloc(cB * cD * 4);
  bf16_t* Hfb   = (bf16_t*)alloc(cB * cD * 2);
  bf16_t* cw1 = (bf16_t*)alloc((size_t)cD * 3 * cD * 2);
  bf16_t* cw2 = (bf16_t*)alloc((size_t)cD * 4 * cD * 2);
  bf16_t* cw3 = (bf16_t*)alloc((size_t)cD * 5 * cD * 2);
  bf16_t* Yb  = (bf16_t*)alloc((size_t)cB * cNY * cD * 2);
  float*  ctx2 = (float*)alloc(cB * cD * 4);
  float*  U2   = gi_f;  // alias: gi_f+gi_b (100.6MB) dead by now; U2 needs 100.5MB
  float*  s2   = (float*)alloc((size_t)cB * cNY * 4);
  float*  Htv  = (float*)alloc(cB * cD * 4);
  bf16_t* Hcat = (bf16_t*)alloc(cB * 2 * cD * 2);
  bf16_t* h1b  = (bf16_t*)alloc(cB * 1024 * 2);
  bf16_t* cls1wb = (bf16_t*)alloc((size_t)1024 * 2 * cD * 2);
  bf16_t* cls2wb = (bf16_t*)alloc((size_t)cC * 1024 * 2);

  auto g1 = [](long long n) { return dim3((unsigned)((n + 255) / 256)); };

  // ---- weight / input conversions ----
  k_f2bf<<<g1(3 * cH * cD), 256, 0, stream>>>(wihf, wihf_b, 3 * cH * cD);
  k_f2bf<<<g1(3 * cH * cD), 256, 0, stream>>>(wihb, wihb_b, 3 * cH * cD);
  k_f2bf<<<g1(cD * cD), 256, 0, stream>>>(W1, W1b, cD * cD);
  k_f2bf<<<g1(cD * cD), 256, 0, stream>>>(W2, W2b, cD * cD);
  k_f2bf<<<g1(cD * cD), 256, 0, stream>>>(W3, W3b, cD * cD);
  k_f2bf<<<g1(cD * cD), 256, 0, stream>>>(W4, W4b, cD * cD);
  k_f2bf<<<g1(cD * 2 * cD), 256, 0, stream>>>(W6w, W6b, cD * 2 * cD);
  k_f2bf<<<g1(cA * cD), 256, 0, stream>>>(lemb, Dmb, cA * cD);
  k_f2bf<<<g1(cA * cA), 256, 0, stream>>>(DDin, DDb, cA * cA);
  k_f2bf<<<g1((long long)cB * cA * cN), 256, 0, stream>>>(DFin, DFb, (long long)cB * cA * cN);
  k_f2bf<<<g1((long long)1024 * 2 * cD), 256, 0, stream>>>(cls1w, cls1wb, (long long)1024 * 2 * cD);
  k_f2bf<<<g1((long long)cC * 1024), 256, 0, stream>>>(cls2w, cls2wb, (long long)cC * 1024);
  k_convw<<<g1(cD * cD * 3), 256, 0, stream>>>(c1w, cw1, 3);
  k_convw<<<g1(cD * cD * 4), 256, 0, stream>>>(c2w, cw2, 4);
  k_convw<<<g1(cD * cD * 5), 256, 0, stream>>>(c3w, cw3, 5);
  k_embed<<<g1((long long)cB * cL * cD), 256, 0, stream>>>(x, wemb, xe_b, (long long)cB * cL * cD);
  k_fm<<<g1((long long)cB * cN * cD), 256, 0, stream>>>(nodes, nmask, eemb, Fmb);
  k_ddsums<<<1, 128, 0, stream>>>(DDin, invRow, invCol);
  k_dfsum<<<g1(cB * cN), 256, 0, stream>>>(DFin, invDF);

  // ---- GRU input projections (WMMA): gi = xe @ wih^T + bih ----
  {
    GemmP p{}; p.A = xe_b; p.sAm = cD; p.sAk = 1; p.sAb = 0;
    p.B = wihf_b; p.sBn = cD; p.sBk = 1;
    p.C = gi_f; p.ldc = 3 * cH; p.sCb = 0;
    p.bias = bihf; p.sBias = 0;
    p.M = cB * cL; p.N = 3 * cH; p.K = cD; p.beta = 0.f;
    launch_gemm(stream, 0, 0, p, 1);
    p.B = wihb_b; p.C = gi_b; p.bias = bihb;
    launch_gemm(stream, 0, 0, p, 1);
  }
  // ---- sequential GRU scans (latency-bound; h in LDS, one WG per batch) ----
  k_gru<<<cB, cH, 0, stream>>>(gi_f, whhf, bhhf, hf, 0);
  k_gru<<<cB, cH, 0, stream>>>(gi_b, whhb, bhhb, hb, 1);
  k_h1<<<g1(cB * cD), 256, 0, stream>>>(hf, hb, H1f, H1b);

  // ---- graph: D_hat = relu(Dm@W1^T + rowScale*(D_D@(Dm@W2^T)) + colScale*(D_D^T@(Dm@W3^T))) ----
  {
    GemmP p{}; p.A = Dmb; p.sAm = cD; p.sAk = 1;
    p.B = W1b; p.sBn = cD; p.sBk = 1;
    p.C = Dhat; p.ldc = cD;
    p.M = cA; p.N = cD; p.K = cD; p.beta = 0.f;
    launch_gemm(stream, 0, 0, p, 1);
    p.B = W2b; p.C = P2; launch_gemm(stream, 0, 0, p, 1);
    p.B = W3b; p.C = P3; launch_gemm(stream, 0, 0, p, 1);
  }
  k_f2bf<<<g1(cA * cD), 256, 0, stream>>>(P2, P2b, cA * cD);
  k_f2bf<<<g1(cA * cD), 256, 0, stream>>>(P3, P3b, cA * cD);
  {
    GemmP p{}; p.A = DDb; p.sAm = cA; p.sAk = 1;       // D_D
    p.B = P2b; p.sBn = 1; p.sBk = cD;                  // (Dm@W2^T)^T as [N,K]
    p.C = Dhat; p.ldc = cD; p.rowScale = invRow;
    p.M = cA; p.N = cD; p.K = cA; p.beta = 1.f;
    launch_gemm(stream, 0, 0, p, 1);
    p.sAm = 1; p.sAk = cA;                              // D_D^T
    p.B = P3b; p.rowScale = invCol;
    launch_gemm(stream, 1, 0, p, 1);                    // + relu
  }
  k_f2bf<<<g1(cA * cD), 256, 0, stream>>>(Dhat, Dhatb, cA * cD);

  // ---- F_hat = relu(Fm@W4^T + invDF*(einsum ban,ad->bnd)) * mask ----
  {
    GemmP p{}; p.A = Fmb; p.sAm = cD; p.sAk = 1; p.sAb = (long long)cN * cD;
    p.B = W4b; p.sBn = cD; p.sBk = 1;
    p.C = Fhat; p.ldc = cD; p.sCb = (long long)cN * cD;
    p.M = cN; p.N = cD; p.K = cD; p.beta = 0.f;
    launch_gemm(stream, 0, 0, p, cB);
  }
  {
    GemmP p{}; p.A = DFb; p.sAm = 1; p.sAk = cN; p.sAb = (long long)cA * cN; // D_F[b]^T
    p.B = Dhatb; p.sBn = 1; p.sBk = cD;                                      // D_hat^T as [N,K]
    p.C = Fhat; p.ldc = cD; p.sCb = (long long)cN * cD;
    p.rowScale = invDF; p.sRS = cN;
    p.postScale = nmask; p.sPS = cN;
    p.M = cN; p.N = cD; p.K = cA; p.beta = 1.f;
    launch_gemm(stream, 1, 0, p, cB);                   // relu then *mask
  }
  k_f2bf<<<g1((long long)cB * cN * cD), 256, 0, stream>>>(Fhat, Fhatb, (long long)cB * cN * cD);

  // ---- attention pool 1 ----
  {
    GemmP p{}; p.A = H1b; p.sAm = cD; p.sAk = 1;
    p.B = W6b + cD; p.sBn = 2 * cD; p.sBk = 1;          // W6b half
    p.C = ctx1; p.ldc = cD; p.bias = W6bias; p.sBias = 0;
    p.M = cB; p.N = cD; p.K = cD; p.beta = 0.f;
    launch_gemm(stream, 0, 0, p, 1);
  }
  {
    GemmP p{}; p.A = Fhatb; p.sAm = cD; p.sAk = 1; p.sAb = (long long)cN * cD;
    p.B = W6b; p.sBn = 2 * cD; p.sBk = 1;               // W6a half
    p.C = U1; p.ldc = cD; p.sCb = (long long)cN * cD;
    p.bias = ctx1; p.sBias = cD;
    p.M = cN; p.N = cD; p.K = cD; p.beta = 0.f;
    launch_gemm(stream, 2, 0, p, cB);                   // tanh
  }
  k_dotv<<<g1((long long)cB * cN), 256, 0, stream>>>(U1, V1, s1, cN);
  k_softmax<<<cB, 256, 0, stream>>>(s1, cN);
  k_wsum<<<g1(cB * cD), 256, 0, stream>>>(s1, U1, Hfv, cN);
  k_f2bf<<<g1(cB * cD), 256, 0, stream>>>(Hfv, Hfb, cB * cD);

  // ---- convs as overlapping-row GEMMs, writing bf16 straight into Y ----
  {
    GemmP p{}; p.A = xe_b; p.sAm = cD; p.sAk = 1; p.sAb = (long long)cL * cD;
    p.ldc = cD; p.sCb = (long long)cNY * cD;
    p.N = cD; p.beta = 0.f; p.sBk = 1;
    p.B = cw1; p.sBn = 3 * cD; p.bias = c1b; p.Cb = Yb;                 p.M = cL - 2; p.K = 3 * cD;
    launch_gemm(stream, 0, 1, p, cB);
    p.B = cw2; p.sBn = 4 * cD; p.bias = c2b; p.Cb = Yb + 2046 * cD;     p.M = cL - 3; p.K = 4 * cD;
    launch_gemm(stream, 0, 1, p, cB);
    p.B = cw3; p.sBn = 5 * cD; p.bias = c3b; p.Cb = Yb + 4091 * cD;     p.M = cL - 4; p.K = 5 * cD;
    launch_gemm(stream, 0, 1, p, cB);
  }

  // ---- attention pool 2 ----
  {
    GemmP p{}; p.A = Hfb; p.sAm = cD; p.sAk = 1;
    p.B = W6b + cD; p.sBn = 2 * cD; p.sBk = 1;
    p.C = ctx2; p.ldc = cD; p.bias = W6bias; p.sBias = 0;
    p.M = cB; p.N = cD; p.K = cD; p.beta = 0.f;
    launch_gemm(stream, 0, 0, p, 1);
  }
  {
    GemmP p{}; p.A = Yb; p.sAm = cD; p.sAk = 1; p.sAb = (long long)cNY * cD;
    p.B = W6b; p.sBn = 2 * cD; p.sBk = 1;
    p.C = U2; p.ldc = cD; p.sCb = (long long)cNY * cD;
    p.bias = ctx2; p.sBias = cD;
    p.M = cNY; p.N = cD; p.K = cD; p.beta = 0.f;
    launch_gemm(stream, 2, 0, p, cB);                   // tanh
  }
  k_dotv<<<g1((long long)cB * cNY), 256, 0, stream>>>(U2, V2, s2, cNY);
  k_softmax<<<cB, 256, 0, stream>>>(s2, cNY);
  k_wsum<<<g1(cB * cD), 256, 0, stream>>>(s2, U2, Htv, cNY);

  // ---- classifier ----
  k_hcat<<<g1(cB * 2 * cD), 256, 0, stream>>>(Hfv, Htv, Hcat);
  {
    GemmP p{}; p.A = Hcat; p.sAm = 2 * cD; p.sAk = 1;
    p.B = cls1wb; p.sBn = 2 * cD; p.sBk = 1;
    p.Cb = h1b; p.ldc = 1024;
    p.bias = cls1b; p.sBias = 0;
    p.M = cB; p.N = 1024; p.K = 2 * cD; p.beta = 0.f;
    launch_gemm(stream, 1, 1, p, 1);                    // relu, bf16 out
  }
  {
    GemmP p{}; p.A = h1b; p.sAm = 1024; p.sAk = 1;
    p.B = cls2wb; p.sBn = 1024; p.sBk = 1;
    p.C = (float*)d_out; p.ldc = cC;
    p.bias = cls2b; p.sBias = 0;
    p.M = cB; p.N = cC; p.K = 1024; p.beta = 0.f;
    launch_gemm(stream, 0, 0, p, 1);
  }
}
